// DCTGCN_63488206569581
// MI455X (gfx1250) — matrix operand order repaired and checked by reading, not compiled, not run on hardware
//
#include <hip/hip_runtime.h>
#include <cstdint>

// ---------------- problem constants ----------------
#define BTOT   4032      // 64*63 batch*windows
#define FDIM   128
#define DDIM   32
#define ODIM   64
#define ME     152       // TWIN*NCHAN
#define MR     56        // TWIN*7 regions
#define NCH    19

typedef __attribute__((ext_vector_type(16))) __bf16 v16bf;
typedef __attribute__((ext_vector_type(8)))  float  v8f;
typedef __attribute__((ext_vector_type(4)))  int    v4i;
typedef __attribute__((address_space(1))) v4i* as1_v4i;   // global
typedef __attribute__((address_space(3))) v4i* as3_v4i;   // LDS

#if __has_builtin(__builtin_amdgcn_global_load_async_to_lds_b128) && \
    __has_builtin(__builtin_amdgcn_s_wait_asynccnt)
#define USE_ASYNC_LDS 1
#else
#define USE_ASYNC_LDS 0
#endif

__constant__ int   c_reg_off[8]  = {0,2,7,9,12,15,17,19};
__constant__ int   c_reg_ch[19]  = {0,1, 3,4,2,5,6, 8,9, 12,13,14, 16,17,18, 7,11, 10,15};
__constant__ int   c_n2r[19]     = {0,0,1,1,1,1,1,5,2,2,6,5,3,3,3,6,4,4,4};
__constant__ float c_rinv[7]     = {0.5f, 0.2f, 0.5f, 1.f/3.f, 1.f/3.f, 0.5f, 0.5f};

// ---------------- bf16 helpers ----------------
__device__ __forceinline__ unsigned short f2bf(float x) {
  union { float f; unsigned u; } c; c.f = x;
  unsigned u = c.u;
  u += 0x7FFFu + ((u >> 16) & 1u);
  return (unsigned short)(u >> 16);
}
__device__ __forceinline__ float bf2f(unsigned short h) {
  union { unsigned u; float f; } c; c.u = ((unsigned)h) << 16; return c.f;
}

union FragU { v16bf v; unsigned short u[16]; };

// A fragment (16x32 bf16): lane&15 = M row, K = hi*16 + i
__device__ __forceinline__ v16bf load_a_frag(const unsigned short* s, int ld, int mrow, int kbase) {
  int lane = threadIdx.x & 31;
  int r  = mrow + (lane & 15);
  int hi = lane >> 4;
  FragU f;
#pragma unroll
  for (int i = 0; i < 16; ++i) {
    int k = kbase + (hi << 4) + i;
    f.u[i] = s[(size_t)r * ld + k];
  }
  return f.v;
}
__device__ __forceinline__ v16bf load_a_frag_g(const unsigned short* s, int ld, int mrow, int kbase, int rmax) {
  int lane = threadIdx.x & 31;
  int r  = mrow + (lane & 15);
  int hi = lane >> 4;
  FragU f;
#pragma unroll
  for (int i = 0; i < 16; ++i) {
    int k = kbase + (hi << 4) + i;
    f.u[i] = (r < rmax) ? s[(size_t)r * ld + k] : (unsigned short)0;
  }
  return f.v;
}
// B fragment (32x16 bf16): lane&15 = N col, K = hi*16 + i, from row-major [K][N]
__device__ __forceinline__ v16bf load_b_frag(const unsigned short* s, int ld, int kbase, int ncol) {
  int lane = threadIdx.x & 31;
  int c  = ncol + (lane & 15);
  int hi = lane >> 4;
  FragU f;
#pragma unroll
  for (int i = 0; i < 16; ++i) {
    int k = kbase + (hi << 4) + i;
    f.u[i] = s[(size_t)k * ld + c];
  }
  return f.v;
}
__device__ __forceinline__ v16bf load_b_frag_g(const unsigned short* s, int ld, int kbase, int ncol, int kmax) {
  int lane = threadIdx.x & 31;
  int c  = ncol + (lane & 15);
  int hi = lane >> 4;
  FragU f;
#pragma unroll
  for (int i = 0; i < 16; ++i) {
    int k = kbase + (hi << 4) + i;
    f.u[i] = (k < kmax) ? s[(size_t)k * ld + c] : (unsigned short)0;
  }
  return f.v;
}
// B fragment from row-major source holding Bᵀ (i.e. B[k][c] = s[c*ld + k])
__device__ __forceinline__ v16bf load_bt_frag(const unsigned short* s, int ld, int kbase, int ncol) {
  int lane = threadIdx.x & 31;
  int c  = ncol + (lane & 15);
  int hi = lane >> 4;
  FragU f;
#pragma unroll
  for (int i = 0; i < 16; ++i) {
    int k = kbase + (hi << 4) + i;
    f.u[i] = s[(size_t)c * ld + k];
  }
  return f.v;
}

__device__ __forceinline__ v8f wmma_bf16(v16bf a, v16bf b, v8f c) {
  return __builtin_amdgcn_wmma_f32_16x16x32_bf16(false, a, false, b, (short)0, c, false, false);
}

// store C/D tile to bf16 row-major (row-guarded)
__device__ __forceinline__ void store_acc_bf16(unsigned short* dst, int ld, int mrow, int ncol, v8f acc, int rmax) {
  int lane = threadIdx.x & 31;
  int col = ncol + (lane & 15);
  int rb  = mrow + ((lane >> 4) << 3);
#pragma unroll
  for (int j = 0; j < 8; ++j) {
    int r = rb + j;
    if (r < rmax) dst[(size_t)r * ld + col] = f2bf(acc[j]);
  }
}
// unguarded store (all rows known in-range)
__device__ __forceinline__ void store_acc_bf16_nog(unsigned short* dst, int ld, int mrow, int ncol, v8f acc) {
  int lane = threadIdx.x & 31;
  int col = ncol + (lane & 15);
  int rb  = mrow + ((lane >> 4) << 3);
#pragma unroll
  for (int j = 0; j < 8; ++j)
    dst[(size_t)(rb + j) * ld + col] = f2bf(acc[j]);
}

// ---------------- small utility kernels ----------------
__global__ void k_zero256(float* p) { p[threadIdx.x] = 0.f; }

__global__ void k_bn_stats(const float* __restrict__ x, float* __restrict__ stats) {
  const int f = threadIdx.x & 127;   // blockDim = 128
  float s = 0.f, ss = 0.f;
  const int nrows = 64 * 256 * NCH;
  for (int row = blockIdx.x; row < nrows; row += gridDim.x) {
    int t = (row / NCH) % 256;
    int a = t - 7;
    int wmin = (a <= 0) ? 0 : ((a + 3) >> 2);
    int wmax = t >> 2; if (wmax > 62) wmax = 62;
    float w = (float)(wmax - wmin + 1);
    float v = x[(size_t)row * FDIM + f];
    s  += w * v;
    ss += w * v * v;
  }
  atomicAdd(&stats[f], s);
  atomicAdd(&stats[FDIM + f], ss);
}

__global__ void k_bn_finalize(const float* __restrict__ stats, const float* __restrict__ gamma,
                              const float* __restrict__ beta, float* __restrict__ ssout) {
  int f = threadIdx.x;  // 128
  const float cnt = (float)((size_t)BTOT * ME);
  float mean = stats[f] / cnt;
  float var  = stats[FDIM + f] / cnt - mean * mean;
  float scale = gamma[f] * rsqrtf(var + 1e-5f);
  ssout[f]        = scale;
  ssout[FDIM + f] = beta[f] - mean * scale;
}

__global__ void k_cvt_bf16(const float* __restrict__ src, unsigned short* __restrict__ dst, int n) {
  for (int i = blockIdx.x * blockDim.x + threadIdx.x; i < n; i += gridDim.x * blockDim.x)
    dst[i] = f2bf(src[i]);
}

// windowed gather + batchnorm -> xn bf16 [BTOT][ME][FDIM]
__global__ void k_build_xn(const float* __restrict__ x, const float* __restrict__ ss,
                           unsigned short* __restrict__ xn) {
  const int total = BTOT * ME * FDIM;
  for (int idx = blockIdx.x * blockDim.x + threadIdx.x; idx < total; idx += gridDim.x * blockDim.x) {
    int f   = idx & 127;
    int rem = idx >> 7;
    int row = rem % ME;
    int bw  = rem / ME;
    int b = bw / 63, w = bw % 63;
    int tau = row / NCH, n = row % NCH;
    int t = w * 4 + tau;
    float v = x[(((size_t)b * 256 + t) * NCH + n) * FDIM + f];
    xn[idx] = f2bf(v * ss[f] + ss[FDIM + f]);
  }
}

// x_reg = Pe @ relu(Q*xn) -> bf16 [BTOT][MR][FDIM]
__global__ void k_build_xr(const unsigned short* __restrict__ xn, const float* __restrict__ Q,
                           unsigned short* __restrict__ xr) {
  const int total = BTOT * MR * FDIM;
  for (int idx = blockIdx.x * blockDim.x + threadIdx.x; idx < total; idx += gridDim.x * blockDim.x) {
    int f   = idx & 127;
    int rem = idx >> 7;
    int r   = rem % MR;
    int bw  = rem / MR;
    int tau = r / 7, rr = r % 7;
    float s = 0.f;
    for (int c = c_reg_off[rr]; c < c_reg_off[rr + 1]; ++c) {
      int n   = c_reg_ch[c];
      int row = tau * NCH + n;
      float v  = bf2f(xn[((size_t)bw * ME + row) * FDIM + f]);
      float qv = Q[(size_t)row * FDIM + f];
      float t  = qv * v;
      s += (t > 0.f) ? t : 0.f;
    }
    xr[idx] = f2bf(s * c_rinv[rr]);
  }
}

// cooperative global->LDS tile stage, async when available
template<int M, int MP>
__device__ __forceinline__ void stage_H_to_lds(unsigned short* Hs, const unsigned short* Hg, int tid) {
#if USE_ASYNC_LDS
  const int chunks = (M * FDIM) / 8;  // 16B per async op
  unsigned short* Hgm = const_cast<unsigned short*>(Hg);
  for (int e = tid; e < chunks; e += 256) {
    __builtin_amdgcn_global_load_async_to_lds_b128(
        (as1_v4i)(Hgm + (size_t)e * 8),
        (as3_v4i)(Hs + (size_t)e * 8),
        0, 0);
  }
  for (int idx = M * FDIM + tid; idx < MP * FDIM; idx += 256) Hs[idx] = 0;
  __builtin_amdgcn_s_wait_asynccnt(0);
#else
  for (int idx = tid; idx < MP * FDIM; idx += 256) {
    int r = idx >> 7;
    Hs[idx] = (r < M) ? Hg[idx] : (unsigned short)0;
  }
#endif
}

// ---------------- WMMA attention: A = softmax((H Wq)(H Wk)ᵀ) ----------------
template<int M, int MP>
__global__ __launch_bounds__(256) void k_attn(const unsigned short* __restrict__ Hbf,
                                              const unsigned short* __restrict__ Wq,
                                              const unsigned short* __restrict__ Wk,
                                              float* __restrict__ Aout) {
  constexpr int NT16 = MP / 16;
  __shared__ unsigned short Hs[MP * FDIM];
  __shared__ unsigned short qs[MP * DDIM];
  __shared__ unsigned short ks2[MP * DDIM];
  const int b   = blockIdx.x;
  const int tid = threadIdx.x;
  const unsigned short* Hg = Hbf + (size_t)b * M * FDIM;
  stage_H_to_lds<M, MP>(Hs, Hg, tid);
  __syncthreads();
  const int wid = __builtin_amdgcn_readfirstlane(tid >> 5);  // wave-uniform
  // q = H@Wq, k = H@Wk  (MPx32 each)
  for (int job = wid; job < NT16 * 4; job += 8) {
    int mt  = job >> 2;
    int mat = (job >> 1) & 1;
    int nt  = job & 1;
    v8f acc = {};
#pragma unroll
    for (int kk = 0; kk < 4; ++kk) {
      v16bf a  = load_a_frag(Hs, FDIM, mt * 16, kk * 32);
      v16bf bb = load_b_frag(mat ? Wk : Wq, DDIM, kk * 32, nt * 16);
      acc = wmma_bf16(a, bb, acc);
    }
    store_acc_bf16_nog(mat ? ks2 : qs, DDIM, mt * 16, nt * 16, acc);
  }
  __syncthreads();
  const int lane  = tid & 31;
  const int cbase = lane & 15;
  const int hi    = lane >> 4;
  for (int mt = wid; mt < NT16; mt += 8) {
    v8f sc[NT16];
    v16bf aq = load_a_frag(qs, DDIM, mt * 16, 0);
#pragma unroll
    for (int nt = 0; nt < NT16; ++nt) {
      v16bf bk = load_bt_frag(ks2, DDIM, 0, nt * 16);
      v8f z = {};
      sc[nt] = wmma_bf16(aq, bk, z);
    }
    float rmax[8], rsum[8], inv[8];
#pragma unroll
    for (int j = 0; j < 8; ++j) rmax[j] = -3.0e38f;
#pragma unroll
    for (int nt = 0; nt < NT16; ++nt) {
      bool valid = (nt * 16 + cbase) < M;
#pragma unroll
      for (int j = 0; j < 8; ++j) {
        float s = valid ? sc[nt][j] : -3.0e38f;
        sc[nt][j] = s;
        rmax[j] = fmaxf(rmax[j], s);
      }
    }
#pragma unroll
    for (int m = 1; m < 16; m <<= 1)
#pragma unroll
      for (int j = 0; j < 8; ++j) rmax[j] = fmaxf(rmax[j], __shfl_xor(rmax[j], m, 32));
#pragma unroll
    for (int j = 0; j < 8; ++j) rsum[j] = 0.f;
#pragma unroll
    for (int nt = 0; nt < NT16; ++nt)
#pragma unroll
      for (int j = 0; j < 8; ++j) {
        float e = __expf(sc[nt][j] - rmax[j]);
        sc[nt][j] = e;
        rsum[j] += e;
      }
#pragma unroll
    for (int m = 1; m < 16; m <<= 1)
#pragma unroll
      for (int j = 0; j < 8; ++j) rsum[j] += __shfl_xor(rsum[j], m, 32);
#pragma unroll
    for (int j = 0; j < 8; ++j) inv[j] = 1.0f / rsum[j];
    float* Ab = Aout + (size_t)b * M * M;
#pragma unroll
    for (int nt = 0; nt < NT16; ++nt) {
      int col = nt * 16 + cbase;
      if (col >= M) continue;
#pragma unroll
      for (int j = 0; j < 8; ++j) {
        int row = mt * 16 + hi * 8 + j;
        if (row < M) Ab[(size_t)row * M + col] = sc[nt][j] * inv[j];
      }
    }
  }
}

// ---------------- O = A @ H  (A f32 from d_out, H/O bf16) ----------------
template<int M, int MP>
__global__ __launch_bounds__(256) void k_bmm(const float* __restrict__ Af,
                                             const unsigned short* __restrict__ Hbf,
                                             unsigned short* __restrict__ Obf) {
  constexpr int NT16 = MP / 16;
  __shared__ unsigned short As[MP * MP];
  const int b   = blockIdx.x;
  const int tid = threadIdx.x;
  const float* Ag = Af + (size_t)b * M * M;
  const unsigned short* Hg = Hbf + (size_t)b * M * FDIM;
  if (tid < 128) __builtin_prefetch(Hg + (size_t)tid * FDIM, 0, 0);  // global_prefetch_b8
  for (int idx = tid; idx < MP * MP; idx += 256) {
    int r = idx / MP, c = idx % MP;
    As[idx] = (r < M && c < M) ? f2bf(Ag[(size_t)r * M + c]) : (unsigned short)0;
  }
  __syncthreads();
  unsigned short* Og = Obf + (size_t)b * M * FDIM;
  const int wid = __builtin_amdgcn_readfirstlane(tid >> 5);
  for (int t = wid; t < NT16 * 8; t += 8) {
    int mt = t >> 3, nt = t & 7;
    v8f acc = {};
#pragma unroll
    for (int kk = 0; kk < MP / 32; ++kk) {
      v16bf a  = load_a_frag(As, MP, mt * 16, kk * 32);
      v16bf bb = load_b_frag_g(Hg, FDIM, kk * 32, nt * 16, M);
      acc = wmma_bf16(a, bb, acc);
    }
    store_acc_bf16(Og, FDIM, mt * 16, nt * 16, acc, M);
  }
}

// ---------------- O = relu(sum_k Hk @ Wk + b) -> bf16 [B][M][ODIM] ----------------
template<int M, int MP, int NT>
__global__ __launch_bounds__(256) void k_gcn(const unsigned short* __restrict__ H0,
                                             const unsigned short* __restrict__ H1,
                                             const unsigned short* __restrict__ H2,
                                             const unsigned short* __restrict__ Wbf,
                                             const float* __restrict__ bias,
                                             unsigned short* __restrict__ Obf) {
  constexpr int NT16 = MP / 16;
  const int b = blockIdx.x;
  const int tid = threadIdx.x, lane = tid & 31;
  const int wid = __builtin_amdgcn_readfirstlane(tid >> 5);
  unsigned short* Og = Obf + (size_t)b * M * ODIM;
  for (int t = wid; t < NT16 * 4; t += 8) {
    int mt = t >> 2, nt = t & 3;
    v8f acc = {};
#pragma unroll
    for (int term = 0; term < NT; ++term) {
      const unsigned short* Hg =
          (term == 0 ? H0 : (term == 1 ? H1 : H2)) + (size_t)b * M * FDIM;
#pragma unroll
      for (int kk = 0; kk < 4; ++kk) {
        v16bf a  = load_a_frag_g(Hg, FDIM, mt * 16, kk * 32, M);
        v16bf bb = load_b_frag(Wbf + (size_t)term * FDIM * ODIM, ODIM, kk * 32, nt * 16);
        acc = wmma_bf16(a, bb, acc);
      }
    }
    int col = nt * 16 + (lane & 15);
    float bv = bias[col];
    int rb = mt * 16 + ((lane >> 4) << 3);
#pragma unroll
    for (int j = 0; j < 8; ++j) {
      int r = rb + j;
      float v = acc[j] + bv;
      v = (v > 0.f) ? v : 0.f;
      if (r < M) Og[(size_t)r * ODIM + col] = f2bf(v);
    }
  }
}

// ---------------- final: X_out = mean_tau( XN + Peᵀ@XRreg ) ----------------
__global__ void k_final(const unsigned short* __restrict__ XNbf,
                        const unsigned short* __restrict__ XRbf,
                        float* __restrict__ Xout) {
  const int total = BTOT * NCH * ODIM;  // [64][63][19][64]
  for (int idx = blockIdx.x * blockDim.x + threadIdx.x; idx < total; idx += gridDim.x * blockDim.x) {
    int f   = idx & 63;
    int rem = idx >> 6;
    int n   = rem % NCH;
    int bw  = rem / NCH;
    int rr  = c_n2r[n];
    float inv = c_rinv[rr];
    float s = 0.f;
#pragma unroll
    for (int tau = 0; tau < 8; ++tau) {
      s += bf2f(XNbf[((size_t)bw * ME + tau * NCH + n) * ODIM + f]);
      s += inv * bf2f(XRbf[((size_t)bw * MR + tau * 7 + rr) * ODIM + f]);
    }
    Xout[idx] = s * 0.125f;
  }
}

// ---------------- launcher ----------------
extern "C" void kernel_launch(void* const* d_in, const int* in_sizes, int n_in,
                              void* d_out, int out_size, void* d_ws, size_t ws_size,
                              hipStream_t stream) {
  const float* x     = (const float*)d_in[0];
  const float* gamma = (const float*)d_in[1];
  const float* beta  = (const float*)d_in[2];
  const float* wq_e  = (const float*)d_in[3];
  const float* wk_e  = (const float*)d_in[4];
  const float* W_E   = (const float*)d_in[5];
  const float* b_E   = (const float*)d_in[6];
  const float* Q     = (const float*)d_in[7];
  const float* wq_r  = (const float*)d_in[8];
  const float* wk_r  = (const float*)d_in[9];
  const float* W_R   = (const float*)d_in[10];
  const float* b_R   = (const float*)d_in[11];

  float* out = (float*)d_out;
  const size_t oX  = 0;
  const size_t oAE = (size_t)64 * 63 * NCH * ODIM;            // 4,902,912
  const size_t oAR = oAE + (size_t)BTOT * ME * ME;            // + 93,155,328
  float* A_E = out + oAE;
  float* A_R = out + oAR;

  char* ws = (char*)d_ws;
  constexpr size_t OFF_STATS = 0;
  constexpr size_t OFF_SS    = 1024;
  constexpr size_t OFF_WQE   = 2048;
  constexpr size_t SZ_QK     = (size_t)FDIM * DDIM * 2;       // 8192 B
  constexpr size_t SZ_WE     = (size_t)3 * FDIM * ODIM * 2;   // 49152 B
  constexpr size_t SZ_WR     = (size_t)2 * FDIM * ODIM * 2;   // 32768 B
  constexpr size_t OFF_WKE   = OFF_WQE + SZ_QK;
  constexpr size_t OFF_WE    = OFF_WKE + SZ_QK;
  constexpr size_t OFF_WQR   = OFF_WE  + SZ_WE;
  constexpr size_t OFF_WKR   = OFF_WQR + SZ_QK;
  constexpr size_t OFF_WR    = OFF_WKR + SZ_QK;
  constexpr size_t OFF_BIG   = 131072;
  constexpr size_t SZ_XN     = (size_t)BTOT * ME * FDIM * 2;  // 156.9 MB
  constexpr size_t SZ_XNOUT  = (size_t)BTOT * ME * ODIM * 2;  //  78.4 MB
  constexpr size_t SZ_XR     = (size_t)BTOT * MR * FDIM * 2;  //  57.8 MB
  constexpr size_t SZ_XROUT  = (size_t)BTOT * MR * ODIM * 2;  //  28.9 MB
  constexpr size_t OFF_XN    = OFF_BIG;
  constexpr size_t OFF_H1    = OFF_XN + SZ_XN;
  constexpr size_t OFF_H2    = OFF_H1 + SZ_XN;
  constexpr size_t OFF_XNOUT = OFF_H2 + SZ_XN;
  constexpr size_t OFF_XRI   = OFF_XNOUT + SZ_XNOUT;
  constexpr size_t OFF_HR1   = OFF_XRI + SZ_XR;
  constexpr size_t OFF_XROUT = OFF_HR1 + SZ_XR;

  float*          stats  = (float*)(ws + OFF_STATS);
  float*          ssbuf  = (float*)(ws + OFF_SS);
  unsigned short* wqe_b  = (unsigned short*)(ws + OFF_WQE);
  unsigned short* wke_b  = (unsigned short*)(ws + OFF_WKE);
  unsigned short* WE_b   = (unsigned short*)(ws + OFF_WE);
  unsigned short* wqr_b  = (unsigned short*)(ws + OFF_WQR);
  unsigned short* wkr_b  = (unsigned short*)(ws + OFF_WKR);
  unsigned short* WR_b   = (unsigned short*)(ws + OFF_WR);
  unsigned short* xn_b   = (unsigned short*)(ws + OFF_XN);
  unsigned short* h1_b   = (unsigned short*)(ws + OFF_H1);
  unsigned short* h2_b   = (unsigned short*)(ws + OFF_H2);
  unsigned short* xnout  = (unsigned short*)(ws + OFF_XNOUT);
  unsigned short* xr_b   = (unsigned short*)(ws + OFF_XRI);
  unsigned short* hr1_b  = (unsigned short*)(ws + OFF_HR1);
  unsigned short* xrout  = (unsigned short*)(ws + OFF_XROUT);

  // 1) batchnorm statistics
  k_zero256<<<1, 256, 0, stream>>>(stats);
  k_bn_stats<<<1024, 128, 0, stream>>>(x, stats);
  k_bn_finalize<<<1, 128, 0, stream>>>(stats, gamma, beta, ssbuf);

  // 2) weight conversion to bf16
  k_cvt_bf16<<<16, 256, 0, stream>>>(wq_e, wqe_b, FDIM * DDIM);
  k_cvt_bf16<<<16, 256, 0, stream>>>(wk_e, wke_b, FDIM * DDIM);
  k_cvt_bf16<<<96, 256, 0, stream>>>(W_E,  WE_b,  3 * FDIM * ODIM);
  k_cvt_bf16<<<16, 256, 0, stream>>>(wq_r, wqr_b, FDIM * DDIM);
  k_cvt_bf16<<<16, 256, 0, stream>>>(wk_r, wkr_b, FDIM * DDIM);
  k_cvt_bf16<<<64, 256, 0, stream>>>(W_R,  WR_b,  2 * FDIM * ODIM);

  // 3) build normalized windows + region pooling
  k_build_xn<<<8192, 256, 0, stream>>>(x, ssbuf, xn_b);
  k_build_xr<<<4096, 256, 0, stream>>>(xn_b, Q, xr_b);

  // 4) attention (WMMA) -> A_E / A_R straight into d_out
  k_attn<ME, 160><<<BTOT, 256, 0, stream>>>(xn_b, wqe_b, wke_b, A_E);
  k_attn<MR, 64><<<BTOT, 256, 0, stream>>>(xr_b, wqr_b, wkr_b, A_R);

  // 5) GCN propagation hops (WMMA batched A@H)
  k_bmm<ME, 160><<<BTOT, 256, 0, stream>>>(A_E, xn_b, h1_b);
  k_bmm<ME, 160><<<BTOT, 256, 0, stream>>>(A_E, h1_b, h2_b);
  k_bmm<MR, 64><<<BTOT, 256, 0, stream>>>(A_R, xr_b, hr1_b);

  // 6) GCN feature combine + bias + relu (WMMA)
  k_gcn<ME, 160, 3><<<BTOT, 256, 0, stream>>>(xn_b, h1_b, h2_b, WE_b, b_E, xnout);
  k_gcn<MR, 64, 2><<<BTOT, 256, 0, stream>>>(xr_b, hr1_b, hr1_b, WR_b, b_R, xrout);

  // 7) expand regions + window mean -> X_out
  k_final<<<4096, 256, 0, stream>>>(xnout, xrout, out + oX);
}
